// TwoSourceAttention_39341900431855
// MI455X (gfx1250) — matrix operand order repaired and checked by reading, compiled, not run on hardware
//
#include <hip/hip_runtime.h>
#include <hip/hip_bf16.h>
#include <cstdint>

// ---------------------------------------------------------------------------
// CDNA5 (gfx1250) wave32 WMMA implementation of the two-source attention block.
// All matmuls run through v_wmma_f32_16x16x32_f16 (f16 inputs, f32 accumulate).
// GEMM: 128x128x32 block tiles, 8 waves, 8 WMMAs per wave per k-step.
// ---------------------------------------------------------------------------

typedef __attribute__((ext_vector_type(16))) _Float16 v16h;
typedef __attribute__((ext_vector_type(8)))  float    v8f;

// Problem constants (match the reference module constants)
constexpr int NB    = 2;
constexpr int NSX   = 1024;
constexpr int NSY   = 1024;
constexpr int ND    = 1024;
constexpr int NH    = 16;
constexpr int NHD   = 64;
constexpr int NFULL = 1536;

// ---------------------------------------------------------------------------
// WMMA helpers
// ---------------------------------------------------------------------------
__device__ inline v8f wmma_f16(v16h a, v16h b, v8f c) {
  // (neg_a, A, neg_b, B, c_mod, C, reuse_a, reuse_b)
  return __builtin_amdgcn_wmma_f32_16x16x32_f16(false, a, false, b, (short)0, c,
                                                false, false);
}

// A operand: 16x32 (MxK) f16 (contiguous k order, matching the B loaders):
//   lanes 0-15  (M=lane): elems 0-15 -> K=0..15
//   lanes 16-31 (M=lane-16): elems 0-15 -> K=16..31
__device__ inline v16h ldA(const _Float16* base, int ld) {
  const int lane = threadIdx.x & 31;
  const int m = lane & 15, hs = lane >> 4;
  v16h a;
#pragma unroll
  for (int i = 0; i < 16; ++i) {
    const int k = i + 16 * hs;
    a[i] = base[m * ld + k];
  }
  return a;
}

// B operand: 32x16 (KxN) f16 stored row-major (k rows, n cols, stride ld).
//   lanes 0-15 (N=lane): K=0..15; lanes 16-31 (N=lane-16): K=16..31
__device__ inline v16h ldB_lin(const _Float16* base, int ld) {
  const int lane = threadIdx.x & 31;
  const int n = lane & 15, kb = (lane >> 4) * 16;
  v16h b;
#pragma unroll
  for (int i = 0; i < 16; ++i) b[i] = base[(kb + i) * ld + n];
  return b;
}

// B operand for S = Q*K^T.  K-dim of WMMA is the head dim d, N-dim is the key.
// Kt is staged in LDS as (key row, d col) with row stride 64.
__device__ inline v16h ldB_kt(const _Float16* Kt, int keyBase, int dBase) {
  const int lane = threadIdx.x & 31;
  const int n = lane & 15, kb = (lane >> 4) * 16;
  v16h b;
#pragma unroll
  for (int i = 0; i < 16; ++i) b[i] = Kt[(keyBase + n) * 64 + dBase + kb + i];
  return b;
}

// B operand for O += P*V.  K-dim is the key (32), N-dim is a 16-wide d chunk.
// Vt staged as (key row, d col) with row stride 64.
__device__ inline v16h ldB_v(const _Float16* Vt, int dBase) {
  const int lane = threadIdx.x & 31;
  const int n = lane & 15, kb = (lane >> 4) * 16;
  v16h b;
#pragma unroll
  for (int i = 0; i < 16; ++i) b[i] = Vt[(kb + i) * 64 + dBase + n];
  return b;
}

// ---------------------------------------------------------------------------
// f32 -> f16 conversion
// ---------------------------------------------------------------------------
__global__ void __launch_bounds__(256) cvt_kernel(const float* __restrict__ in,
                                                  _Float16* __restrict__ out,
                                                  long long n) {
  long long i = (long long)blockIdx.x * blockDim.x + threadIdx.x;
  const long long stride = (long long)gridDim.x * blockDim.x;
  for (; i < n; i += stride) out[i] = (_Float16)in[i];
}

// ---------------------------------------------------------------------------
// Tiled WMMA GEMM:  C[MxN] = A[MxK](f16,row-major) * B[KxN](f16,row-major) + bias
// 256 threads = 8 waves; block tile 128x128, BK=32.  Waves form a 4x2 grid;
// each wave computes a 32x64 strip: 2 A-operands x 4 B-operands = 8 WMMAs/step
// into 8 persistent f32 accumulators.
// ---------------------------------------------------------------------------
template <typename OutT>
__global__ void __launch_bounds__(256) gemm_wmma_kernel(
    const _Float16* __restrict__ A, const _Float16* __restrict__ Bm,
    const float* __restrict__ bias, OutT* __restrict__ C,
    int M, int N, int K) {
  __shared__ __align__(16) _Float16 As[128 * 32];
  __shared__ __align__(16) _Float16 Bs[32 * 128];

  const int tid  = threadIdx.x;
  const int lane = tid & 31;
  const int wave = tid >> 5;
  const int wm = wave >> 1;   // 0..3 -> 32-row strip
  const int wn = wave & 1;    // 0..1 -> 64-col strip
  const int tm = blockIdx.x * 128;
  const int tn = blockIdx.y * 128;

  v8f acc[2][4] = {};

  for (int k0 = 0; k0 < K; k0 += 32) {
    // Cooperative staging: each thread moves 32 B of A and 32 B of B (2x b128).
    {
      const int idx = tid * 16;
      const int ra = idx >> 5, ca = idx & 31;    // A: 128 rows x 32 cols
      const _Float16* ag = &A[(size_t)(tm + ra) * K + (k0 + ca)];
      *(uint4*)&As[idx]     = *(const uint4*)ag;
      *(uint4*)&As[idx + 8] = *(const uint4*)(ag + 8);
      const int rb = idx >> 7, cb = idx & 127;   // B: 32 rows x 128 cols
      const _Float16* bg = &Bm[(size_t)(k0 + rb) * N + (tn + cb)];
      *(uint4*)&Bs[idx]     = *(const uint4*)bg;
      *(uint4*)&Bs[idx + 8] = *(const uint4*)(bg + 8);
    }
    __syncthreads();

    const v16h a0 = ldA(&As[(wm * 32 + 0) * 32], 32);
    const v16h a1 = ldA(&As[(wm * 32 + 16) * 32], 32);
#pragma unroll
    for (int j = 0; j < 4; ++j) {
      const v16h b = ldB_lin(&Bs[wn * 64 + j * 16], 128);
      acc[0][j] = wmma_f16(a0, b, acc[0][j]);
      acc[1][j] = wmma_f16(a1, b, acc[1][j]);
    }
    __syncthreads();
  }

  // Store D (C-matrix layout: row m = vi + 8*(lane>=16), col n = lane&15)
  const int n  = lane & 15;
  const int hs = lane >> 4;
#pragma unroll
  for (int j = 0; j < 4; ++j) {
    const int col = tn + wn * 64 + j * 16 + n;
    const float bv = bias[col];
#pragma unroll
    for (int r = 0; r < 2; ++r) {
#pragma unroll
      for (int vi = 0; vi < 8; ++vi) {
        const int row = tm + wm * 32 + r * 16 + vi + 8 * hs;
        C[(size_t)row * N + col] = (OutT)(acc[r][j][vi] + bv);
      }
    }
  }
}

// ---------------------------------------------------------------------------
// Flash self-attention (causal + key mask), one wave per (b, h, 16-query tile).
// qkv: (B, SX, 3D) f16 ; maskx: (B,1,1,SX) f32 ; writes f16 into cat[:, :, 0:D]
// ---------------------------------------------------------------------------
__global__ void __launch_bounds__(32) flash_self_kernel(
    const _Float16* __restrict__ qkv, const float* __restrict__ maskx,
    _Float16* __restrict__ cat) {
  const int qt = blockIdx.x, h = blockIdx.y, b = blockIdx.z;
  const int lane = threadIdx.x;
  const int q0row = qt * 16;
  const float scale = 0.125f;  // 1/sqrt(64)

  __shared__ __align__(16) _Float16 Kt[32 * 64];
  __shared__ __align__(16) _Float16 Vt[32 * 64];
  __shared__ float    Sb[16 * 32];
  __shared__ _Float16 Pb[16 * 32];
  __shared__ float    rs[16];

  // Q as two A operands (d 0..31 and 32..63)
  v16h qa0, qa1;
  {
    const int m = lane & 15, hs = lane >> 4;
    const _Float16* qp =
        qkv + ((size_t)(b * NSX + q0row + m)) * (3 * ND) + h * NHD;
#pragma unroll
    for (int i = 0; i < 16; ++i) {
      const int k = i + 16 * hs;
      qa0[i] = qp[k];
      qa1[i] = qp[32 + k];
    }
  }

  v8f o0 = {}, o1 = {}, o2 = {}, o3 = {};
  float mrow = -3.0e38f, lrow = 0.0f;

  const int qhi = q0row + 15;
  const int nkb = (qhi >> 5) + 1;   // 32-key blocks up to the causal frontier

  for (int kb = 0; kb < nkb; ++kb) {
    const int ks = kb * 32;
    // Stage K and V rows (one row of 64 halves per lane, 16-B vector moves)
    {
      const int krow = ks + lane;
      uint4* kd = (uint4*)&Kt[lane * 64];
      uint4* vd = (uint4*)&Vt[lane * 64];
      const _Float16* kp =
          qkv + ((size_t)(b * NSX + krow)) * (3 * ND) + ND + h * NHD;
      const _Float16* vp = kp + ND;
#pragma unroll
      for (int i = 0; i < 8; ++i) {
        kd[i] = ((const uint4*)kp)[i];
        vd[i] = ((const uint4*)vp)[i];
      }
    }
    __syncthreads();

    // S = Q * K^T for 16 q x 32 k  (two C tiles, two d-chunks each)
    v8f s0 = {}, s1 = {};
    s0 = wmma_f16(qa0, ldB_kt(Kt, 0, 0),  s0);
    s0 = wmma_f16(qa1, ldB_kt(Kt, 0, 32), s0);
    s1 = wmma_f16(qa0, ldB_kt(Kt, 16, 0),  s1);
    s1 = wmma_f16(qa1, ldB_kt(Kt, 16, 32), s1);
    {
      const int n = lane & 15, hs = lane >> 4;
#pragma unroll
      for (int vi = 0; vi < 8; ++vi) {
        const int m = vi + 8 * hs;
        Sb[m * 32 + n]      = s0[vi];
        Sb[m * 32 + 16 + n] = s1[vi];
      }
    }
    __syncthreads();

    // Online softmax: lane r (<16) owns query row q0row+r
    if (lane < 16) {
      const int qi = q0row + lane;
      const float mOld = mrow;
      float mNew = mOld;
      float sv[32];
#pragma unroll
      for (int j = 0; j < 32; ++j) {
        const int jj = ks + j;
        float s = Sb[lane * 32 + j];
        const float mx = maskx[b * NSX + jj];
        s = (jj <= qi) ? (s * scale + mx) : (-10000.0f + mx);
        sv[j] = s;
        mNew = fmaxf(mNew, s);
      }
      const float f = __expf(mOld - mNew);
      float ls = 0.0f;
#pragma unroll
      for (int j = 0; j < 32; ++j) {
        const float p = __expf(sv[j] - mNew);
        ls += p;
        Pb[lane * 32 + j] = (_Float16)p;
      }
      lrow = lrow * f + ls;
      mrow = mNew;
      rs[lane] = f;
    }
    __syncthreads();

    // Rescale accumulators by per-row factor, then O += P*V
    {
      const int hs = lane >> 4;
      float fv[8];
#pragma unroll
      for (int vi = 0; vi < 8; ++vi) fv[vi] = rs[vi + 8 * hs];
#pragma unroll
      for (int vi = 0; vi < 8; ++vi) {
        o0[vi] *= fv[vi]; o1[vi] *= fv[vi];
        o2[vi] *= fv[vi]; o3[vi] *= fv[vi];
      }
    }
    const v16h pa = ldA(Pb, 32);
    o0 = wmma_f16(pa, ldB_v(Vt, 0),  o0);
    o1 = wmma_f16(pa, ldB_v(Vt, 16), o1);
    o2 = wmma_f16(pa, ldB_v(Vt, 32), o2);
    o3 = wmma_f16(pa, ldB_v(Vt, 48), o3);
    __syncthreads();
  }

  if (lane < 16) rs[lane] = 1.0f / lrow;
  __syncthreads();
  {
    const int n = lane & 15, hs = lane >> 4;
    _Float16* op = cat + ((size_t)(b * NSX + q0row)) * (2 * ND) + h * NHD;
#pragma unroll
    for (int vi = 0; vi < 8; ++vi) {
      const int m = vi + 8 * hs;
      const float inv = rs[m];
      op[(size_t)m * (2 * ND) + n]      = (_Float16)(o0[vi] * inv);
      op[(size_t)m * (2 * ND) + 16 + n] = (_Float16)(o1[vi] * inv);
      op[(size_t)m * (2 * ND) + 32 + n] = (_Float16)(o2[vi] * inv);
      op[(size_t)m * (2 * ND) + 48 + n] = (_Float16)(o3[vi] * inv);
    }
  }
}

// ---------------------------------------------------------------------------
// Flash cross-attention in gathered coordinates:
//   query i of batch b sits at qpos = FULL - len_x[b] + i
//   keys j in [0, len_y[b]) allowed iff j <= qpos.
// q2: (B,SX,D) f16 ; kv2: (B,SY,2D) f16 (k2 cols [0,D), v2 cols [D,2D))
// Writes f16 into cat[:, :, D:2D], zero for rows >= len_x[b].
// ---------------------------------------------------------------------------
__global__ void __launch_bounds__(32) flash_cross_kernel(
    const _Float16* __restrict__ q2, const _Float16* __restrict__ kv2,
    _Float16* __restrict__ cat) {
  const int qt = blockIdx.x, h = blockIdx.y, b = blockIdx.z;
  const int lane = threadIdx.x;
  const int LX = (b == 0) ? 1024 : 700;
  const int LY = (b == 0) ? 1024 : 850;
  const int off = NFULL - LX;
  const int q0row = qt * 16;
  const float scale = 0.125f;

  __shared__ __align__(16) _Float16 Kt[32 * 64];
  __shared__ __align__(16) _Float16 Vt[32 * 64];
  __shared__ float    Sb[16 * 32];
  __shared__ _Float16 Pb[16 * 32];
  __shared__ float    rs[16];

  v16h qa0, qa1;
  {
    const int m = lane & 15, hs = lane >> 4;
    const _Float16* qp = q2 + ((size_t)(b * NSX + q0row + m)) * ND + h * NHD;
#pragma unroll
    for (int i = 0; i < 16; ++i) {
      const int k = i + 16 * hs;
      qa0[i] = qp[k];
      qa1[i] = qp[32 + k];
    }
  }

  v8f o0 = {}, o1 = {}, o2 = {}, o3 = {};
  float mrow = -3.0e38f, lrow = 0.0f;

  int nkb = 0;
  if (q0row < LX) {
    const int qposmax = off + q0row + 15;
    const int jlim = (LY < qposmax + 1) ? LY : (qposmax + 1);
    nkb = (jlim + 31) >> 5;
  }

  for (int kb = 0; kb < nkb; ++kb) {
    const int ks = kb * 32;
    {
      const int krow = ks + lane;
      uint4* kd = (uint4*)&Kt[lane * 64];
      uint4* vd = (uint4*)&Vt[lane * 64];
      if (krow < NSY) {
        const _Float16* kp = kv2 + ((size_t)(b * NSY + krow)) * (2 * ND) + h * NHD;
        const _Float16* vp = kp + ND;
#pragma unroll
        for (int i = 0; i < 8; ++i) {
          kd[i] = ((const uint4*)kp)[i];
          vd[i] = ((const uint4*)vp)[i];
        }
      } else {
        const uint4 z = {0u, 0u, 0u, 0u};
#pragma unroll
        for (int i = 0; i < 8; ++i) { kd[i] = z; vd[i] = z; }
      }
    }
    __syncthreads();

    v8f s0 = {}, s1 = {};
    s0 = wmma_f16(qa0, ldB_kt(Kt, 0, 0),  s0);
    s0 = wmma_f16(qa1, ldB_kt(Kt, 0, 32), s0);
    s1 = wmma_f16(qa0, ldB_kt(Kt, 16, 0),  s1);
    s1 = wmma_f16(qa1, ldB_kt(Kt, 16, 32), s1);
    {
      const int n = lane & 15, hs = lane >> 4;
#pragma unroll
      for (int vi = 0; vi < 8; ++vi) {
        const int m = vi + 8 * hs;
        Sb[m * 32 + n]      = s0[vi];
        Sb[m * 32 + 16 + n] = s1[vi];
      }
    }
    __syncthreads();

    if (lane < 16) {
      const int qpos = off + q0row + lane;
      const float mOld = mrow;
      float mNew = mOld;
      float sv[32];
#pragma unroll
      for (int j = 0; j < 32; ++j) {
        const int jj = ks + j;
        float s = Sb[lane * 32 + j];
        const bool ok = (jj <= qpos) && (jj < LY);
        s = ok ? (s * scale) : -10000.0f;
        sv[j] = s;
        mNew = fmaxf(mNew, s);
      }
      const float f = __expf(mOld - mNew);
      float ls = 0.0f;
#pragma unroll
      for (int j = 0; j < 32; ++j) {
        const float p = __expf(sv[j] - mNew);
        ls += p;
        Pb[lane * 32 + j] = (_Float16)p;
      }
      lrow = lrow * f + ls;
      mrow = mNew;
      rs[lane] = f;
    }
    __syncthreads();

    {
      const int hs = lane >> 4;
      float fv[8];
#pragma unroll
      for (int vi = 0; vi < 8; ++vi) fv[vi] = rs[vi + 8 * hs];
#pragma unroll
      for (int vi = 0; vi < 8; ++vi) {
        o0[vi] *= fv[vi]; o1[vi] *= fv[vi];
        o2[vi] *= fv[vi]; o3[vi] *= fv[vi];
      }
    }
    const v16h pa = ldA(Pb, 32);
    o0 = wmma_f16(pa, ldB_v(Vt, 0),  o0);
    o1 = wmma_f16(pa, ldB_v(Vt, 16), o1);
    o2 = wmma_f16(pa, ldB_v(Vt, 32), o2);
    o3 = wmma_f16(pa, ldB_v(Vt, 48), o3);
    __syncthreads();
  }

  if (lane < 16) rs[lane] = (lrow > 0.0f) ? (1.0f / lrow) : 0.0f;
  __syncthreads();
  {
    const int n = lane & 15, hs = lane >> 4;
    _Float16* op = cat + ((size_t)(b * NSX + q0row)) * (2 * ND) + ND + h * NHD;
#pragma unroll
    for (int vi = 0; vi < 8; ++vi) {
      const int m = vi + 8 * hs;
      const bool act = (q0row + m) < LX;
      const float inv = rs[m];
      op[(size_t)m * (2 * ND) + n]      = (_Float16)(act ? o0[vi] * inv : 0.0f);
      op[(size_t)m * (2 * ND) + 16 + n] = (_Float16)(act ? o1[vi] * inv : 0.0f);
      op[(size_t)m * (2 * ND) + 32 + n] = (_Float16)(act ? o2[vi] * inv : 0.0f);
      op[(size_t)m * (2 * ND) + 48 + n] = (_Float16)(act ? o3[vi] * inv : 0.0f);
    }
  }
}

// ---------------------------------------------------------------------------
// Host-side orchestration
// ---------------------------------------------------------------------------
extern "C" void kernel_launch(void* const* d_in, const int* in_sizes, int n_in,
                              void* d_out, int out_size, void* d_ws,
                              size_t ws_size, hipStream_t stream) {
  (void)in_sizes; (void)n_in; (void)out_size; (void)ws_size;

  const float* x      = (const float*)d_in[0];
  const float* y      = (const float*)d_in[1];
  const float* maskx  = (const float*)d_in[2];
  const float* W_attn = (const float*)d_in[3];
  const float* b_attn = (const float*)d_in[4];
  const float* W_2a   = (const float*)d_in[5];
  const float* b_2a   = (const float*)d_in[6];
  const float* W_2b   = (const float*)d_in[7];
  const float* b_2b   = (const float*)d_in[8];
  const float* W_proj = (const float*)d_in[9];
  const float* b_proj = (const float*)d_in[10];

  char* ws = (char*)d_ws;
  const size_t MB = 1ull << 20;
  _Float16* x_h   = (_Float16*)(ws + 0 * MB);   //  4 MiB : (B,SX,D) f16
  _Float16* y_h   = (_Float16*)(ws + 4 * MB);   //  4 MiB
  _Float16* Wa_h  = (_Float16*)(ws + 8 * MB);   //  6 MiB : (D,3D)
  _Float16* W2a_h = (_Float16*)(ws + 14 * MB);  //  2 MiB : (D,D)
  _Float16* W2b_h = (_Float16*)(ws + 16 * MB);  //  4 MiB : (D,2D)
  _Float16* Wp_h  = (_Float16*)(ws + 20 * MB);  //  4 MiB : (2D,D)
  _Float16* qkv_h = (_Float16*)(ws + 24 * MB);  // 12 MiB : (B,SX,3D)
  _Float16* q2_h  = (_Float16*)(ws + 36 * MB);  //  4 MiB : (B,SX,D)
  _Float16* kv2_h = (_Float16*)(ws + 40 * MB);  //  8 MiB : (B,SY,2D)
  _Float16* cat_h = (_Float16*)(ws + 48 * MB);  //  8 MiB : (B,SX,2D)

  const int M = NB * NSX;  // 2048

  // f32 -> f16 conversions
  cvt_kernel<<<512, 256, 0, stream>>>(x, x_h, (long long)M * ND);
  cvt_kernel<<<512, 256, 0, stream>>>(y, y_h, (long long)NB * NSY * ND);
  cvt_kernel<<<512, 256, 0, stream>>>(W_attn, Wa_h, (long long)ND * 3 * ND);
  cvt_kernel<<<512, 256, 0, stream>>>(W_2a, W2a_h, (long long)ND * ND);
  cvt_kernel<<<512, 256, 0, stream>>>(W_2b, W2b_h, (long long)ND * 2 * ND);
  cvt_kernel<<<512, 256, 0, stream>>>(W_proj, Wp_h, (long long)2 * ND * ND);

  // qkv = x @ W_attn + b_attn         (2048 x 3072, K=1024)
  gemm_wmma_kernel<_Float16><<<dim3(M / 128, (3 * ND) / 128), 256, 0, stream>>>(
      x_h, Wa_h, b_attn, qkv_h, M, 3 * ND, ND);
  // q2 = x @ W_2a + b_2a              (2048 x 1024, K=1024)
  gemm_wmma_kernel<_Float16><<<dim3(M / 128, ND / 128), 256, 0, stream>>>(
      x_h, W2a_h, b_2a, q2_h, M, ND, ND);
  // k2v2 = y @ W_2b + b_2b            (2048 x 2048, K=1024)
  gemm_wmma_kernel<_Float16><<<dim3(M / 128, (2 * ND) / 128), 256, 0, stream>>>(
      y_h, W2b_h, b_2b, kv2_h, M, 2 * ND, ND);

  // attention heads: one wave per (q-tile, head, batch)
  flash_self_kernel<<<dim3(NSX / 16, NH, NB), 32, 0, stream>>>(qkv_h, maskx,
                                                               cat_h);
  flash_cross_kernel<<<dim3(NSX / 16, NH, NB), 32, 0, stream>>>(q2_h, kv2_h,
                                                                cat_h);

  // out = concat(a, a2m) @ W_proj + b_proj   (2048 x 1024, K=2048) -> f32
  gemm_wmma_kernel<float><<<dim3(M / 128, ND / 128), 256, 0, stream>>>(
      cat_h, Wp_h, b_proj, (float*)d_out, M, ND, 2 * ND);
}
